// CosfacePairwiseLoss_37623913513784
// MI455X (gfx1250) — matrix operand order repaired and checked by reading, not compiled, not run on hardware
//
#include <hip/hip_runtime.h>
#include <hip/hip_bf16.h>
#include <hip/hip_fp16.h>

// ---------------------------------------------------------------------------
// CosFace pairwise loss, fused:  loss = mean_i softplus(log Σ_pos + log Σ_neg)
//   sim = (feat/||feat||) @ (feat/||feat||)^T   (8192x8192, K=512)
//   Σ_pos(i) = Σ_{j: lab_i==lab_j} exp(30*(0.3 - sim_ij))
//   Σ_neg(i) = Σ_{j: lab_i!=lab_j} exp(30*sim_ij)
// (NEG_FILL=-10 terms underflow to 0 in fp32; |30*s| <= 300 so no max-shift.)
//
// Main kernel: flash-style fusion, f16 WMMA with f32 accum, async double-
// buffered B tiles (GLOBAL_LOAD_ASYNC_TO_LDS_B128 / ASYNCcnt), software-
// pipelined fragment loads, 16x32 output per wave (A-fragment reuse).
// ---------------------------------------------------------------------------

typedef _Float16 v4h  __attribute__((ext_vector_type(4)));
typedef _Float16 v8h  __attribute__((ext_vector_type(8)));
typedef _Float16 v16h __attribute__((ext_vector_type(16)));
typedef float    v8f  __attribute__((ext_vector_type(8)));

#define N_ROWS        8192
#define K_DIM         512
#define LDS_K         (K_DIM + 8)   // padded stride (halves): 1040B rows, conflict-light
#define ROWS_PER_BLK  64
#define COLS_PER_TILE 64            // B tile width (columns) per pipeline stage
#define COLS_PER_BLK  1024
#define TILE_ITERS    (COLS_PER_BLK / COLS_PER_TILE)
#define K2            43.28085122666891f   // 30 / ln(2):  exp(30*x) = exp2(x*K2)
#define MARGIN        0.3f

// ---------------------------------------------------------------------------
__global__ __launch_bounds__(256) void init_kernel(float* sums, float* out) {
  int i = blockIdx.x * 256 + threadIdx.x;
  if (i < 2 * N_ROWS) sums[i] = 0.0f;
  if (i == 0) out[0] = 0.0f;
}

// ---------------------------------------------------------------------------
// One wave32 per row: sum-of-squares reduce, clamp, scale, convert to f16.
__global__ __launch_bounds__(256) void normalize_kernel(
    const float* __restrict__ feat, _Float16* __restrict__ fH) {
  const int wave = threadIdx.x >> 5;
  const int lane = threadIdx.x & 31;
  const size_t row = (size_t)blockIdx.x * 8 + wave;

  const float4* src = (const float4*)(feat + row * K_DIM);
  float4 v[4];
  float ss = 0.0f;
#pragma unroll
  for (int i = 0; i < 4; ++i) {
    v[i] = src[lane + 32 * i];
    ss = fmaf(v[i].x, v[i].x, ss);
    ss = fmaf(v[i].y, v[i].y, ss);
    ss = fmaf(v[i].z, v[i].z, ss);
    ss = fmaf(v[i].w, v[i].w, ss);
  }
#pragma unroll
  for (int off = 16; off; off >>= 1) ss += __shfl_xor(ss, off, 32);
  const float inv = 1.0f / fmaxf(sqrtf(ss), 1e-8f);

  v4h* dst = (v4h*)(fH + row * K_DIM);
#pragma unroll
  for (int i = 0; i < 4; ++i) {
    v4h h;
    h[0] = (_Float16)(v[i].x * inv);
    h[1] = (_Float16)(v[i].y * inv);
    h[2] = (_Float16)(v[i].z * inv);
    h[3] = (_Float16)(v[i].w * inv);
    dst[lane + 32 * i] = h;
  }
}

// ---------------------------------------------------------------------------
// Fragment loaders from row-major f16 LDS tiles (contiguous k = 16g + e per lane half, same for A and B).
// A 16x32: lane(m=l&15,g=l>>4): elems 0-15 <- K[16g .. 16g+15] of row m
__device__ __forceinline__ v16h load_fragA(const _Float16* tile, int lane, int kk) {
  const int m = lane & 15, g = lane >> 4;
  const _Float16* p = tile + m * LDS_K + kk + (g << 4);
  v8h lo = *(const v8h*)(p);
  v8h hi = *(const v8h*)(p + 8);
  v16h f;
#pragma unroll
  for (int e = 0; e < 8; ++e) { f[e] = lo[e]; f[e + 8] = hi[e]; }
  return f;
}
// B 32x16: lane(n=l&15,g=l>>4): elems 0-15 <- K[16g .. 16g+15] of column n
__device__ __forceinline__ v16h load_fragB(const _Float16* tile, int lane, int kk) {
  const int n = lane & 15, g = lane >> 4;
  const _Float16* p = tile + n * LDS_K + kk + (g << 4);
  v8h lo = *(const v8h*)(p);
  v8h hi = *(const v8h*)(p + 8);
  v16h f;
#pragma unroll
  for (int e = 0; e < 8; ++e) { f[e] = lo[e]; f[e + 8] = hi[e]; }
  return f;
}

// ---------------------------------------------------------------------------
// Async-stage one 64-col x 512-K f16 tile into LDS (tracked by ASYNCcnt).
// 4096 b128 transfers, 16 per thread.  VDST = 32-bit LDS byte address,
// VADDR = 64-bit global address ("GV" mode, saddr=off).
__device__ __forceinline__ void stage_async_tile(
    _Float16* __restrict__ buf, const _Float16* __restrict__ src) {
  for (int v = threadIdx.x; v < COLS_PER_TILE * (K_DIM / 8); v += 256) {
    const int rr = v >> 6;            // /(512/8)
    const int kc = (v & 63) << 3;
    const unsigned           lds = (unsigned)(size_t)&buf[rr * LDS_K + kc];
    const unsigned long long gaddr =
        (unsigned long long)(size_t)&src[(size_t)rr * K_DIM + kc];
    asm volatile("global_load_async_to_lds_b128 %0, %1, off"
                 :: "v"(lds), "v"(gaddr) : "memory");
  }
}
__device__ __forceinline__ void wait_async0() {
  asm volatile("s_wait_asynccnt 0x0" ::: "memory");
}

// ---------------------------------------------------------------------------
// Fused GEMM + masked-exp row reduction.
// Grid: (N_ROWS/ROWS_PER_BLK, N_ROWS/COLS_PER_BLK). Block: 256 = 8 waves as
// 4 (M-tiles of 16) x 2 (N-halves of 32); each wave outputs 16x32 per stage.
__global__ __launch_bounds__(256) void cosface_main_kernel(
    const _Float16* __restrict__ fH, const int* __restrict__ label,
    float* __restrict__ gSumP, float* __restrict__ gSumN) {
  extern __shared__ _Float16 smem[];
  _Float16* Alds  = smem;                                   // 64 * LDS_K
  _Float16* Blds0 = smem + ROWS_PER_BLK * LDS_K;            // 64 * LDS_K
  _Float16* Blds1 = Blds0 + COLS_PER_TILE * LDS_K;          // 64 * LDS_K

  const int tid  = threadIdx.x;
  const int wave = tid >> 5;
  const int lane = tid & 31;
  const int mT   = wave >> 1;   // 0..3  -> 16-row tile
  const int nS   = wave & 1;    // 0..1  -> 32-col half of the B tile
  const int lidx = lane & 15;
  const int g    = lane >> 4;

  const int row0    = blockIdx.x * ROWS_PER_BLK;
  const int colBase = blockIdx.y * COLS_PER_BLK;

  // Stage A tile (64 rows x 512) once (synchronous vector copies).
  for (int v = tid; v < ROWS_PER_BLK * (K_DIM / 8); v += 256) {
    const int rr = v >> 6;
    const int kc = (v & 63) << 3;
    *(v8h*)&Alds[rr * LDS_K + kc] =
        *(const v8h*)&fH[(size_t)(row0 + rr) * K_DIM + kc];
  }

  // Prologue: kick off async staging of B tile 0 into buffer 0.
  stage_async_tile(Blds0, fH + (size_t)colBase * K_DIM);

  // This wave's 16 row labels (C layout: m = r + 8*g).
  int labR[8];
#pragma unroll
  for (int r = 0; r < 8; ++r)
    labR[r] = label[row0 + mT * 16 + r + 8 * g];

  float sumP[8], sumN[8];
#pragma unroll
  for (int r = 0; r < 8; ++r) { sumP[r] = 0.0f; sumN[r] = 0.0f; }

  const _Float16* At = Alds + mT * 16 * LDS_K;

  for (int it = 0; it < TILE_ITERS; ++it) {
    wait_async0();        // this wave's async writes to buf[it&1] are in LDS
    __syncthreads();      // all waves staged buf[it&1]; prior reads of the
                          // other buffer are complete -> safe to overwrite it
    if (it + 1 < TILE_ITERS)
      stage_async_tile((it & 1) ? Blds0 : Blds1,
                       fH + (size_t)(colBase + (it + 1) * COLS_PER_TILE) * K_DIM);

    const _Float16* Bt  = ((it & 1) ? Blds1 : Blds0) + nS * 32 * LDS_K;
    const int j0   = colBase + it * COLS_PER_TILE + nS * 32 + lidx;
    const int labC0 = label[j0];
    const int labC1 = label[j0 + 16];

    v8f acc0 = {0.f, 0.f, 0.f, 0.f, 0.f, 0.f, 0.f, 0.f};
    v8f acc1 = {0.f, 0.f, 0.f, 0.f, 0.f, 0.f, 0.f, 0.f};

    // Software-pipelined K loop: fetch k+1 fragments before consuming k.
    v16h a  = load_fragA(At, lane, 0);
    v16h b0 = load_fragB(Bt, lane, 0);
    v16h b1 = load_fragB(Bt + 16 * LDS_K, lane, 0);
#pragma unroll
    for (int kk = 0; kk < K_DIM; kk += 32) {
      const v16h ca = a, cb0 = b0, cb1 = b1;
      if (kk + 32 < K_DIM) {
        a  = load_fragA(At, lane, kk + 32);
        b0 = load_fragB(Bt, lane, kk + 32);
        b1 = load_fragB(Bt + 16 * LDS_K, lane, kk + 32);
      }
      acc0 = __builtin_amdgcn_wmma_f32_16x16x32_f16(
          false, ca, false, cb0, (short)0, acc0, false, false);
      acc1 = __builtin_amdgcn_wmma_f32_16x16x32_f16(
          false, ca, false, cb1, (short)0, acc1, false, false);
    }

    // Epilogue: masked exp accumulation (lane covers rows m=r+8g, cols j0,j0+16).
#pragma unroll
    for (int r = 0; r < 8; ++r) {
      const float s0 = acc0[r];
      const float s1 = acc1[r];
      if (labR[r] == labC0) sumP[r] += exp2f((MARGIN - s0) * K2);
      else                  sumN[r] += exp2f(s0 * K2);
      if (labR[r] == labC1) sumP[r] += exp2f((MARGIN - s1) * K2);
      else                  sumN[r] += exp2f(s1 * K2);
    }
  }

  // Reduce across the 16 lanes sharing the same 8 rows, then global atomics.
#pragma unroll
  for (int r = 0; r < 8; ++r) {
    float p = sumP[r], n = sumN[r];
#pragma unroll
    for (int off = 8; off; off >>= 1) {
      p += __shfl_xor(p, off, 16);
      n += __shfl_xor(n, off, 16);
    }
    if (lidx == 0) {
      const int row = row0 + mT * 16 + r + 8 * g;
      atomicAdd(&gSumP[row], p);
      atomicAdd(&gSumN[row], n);
    }
  }
}

// ---------------------------------------------------------------------------
__global__ __launch_bounds__(256) void finalize_kernel(
    const float* __restrict__ gSumP, const float* __restrict__ gSumN,
    float* __restrict__ out) {
  const int i = blockIdx.x * 256 + threadIdx.x;
  const float z = __logf(gSumP[i]) + __logf(gSumN[i]);
  float v = (z > 20.0f) ? z : log1pf(__expf(z));   // softplus
#pragma unroll
  for (int off = 16; off; off >>= 1) v += __shfl_xor(v, off, 32);

  __shared__ float red[8];
  const int wave = threadIdx.x >> 5, lane = threadIdx.x & 31;
  if (lane == 0) red[wave] = v;
  __syncthreads();
  if (threadIdx.x == 0) {
    float t = 0.0f;
#pragma unroll
    for (int w = 0; w < 8; ++w) t += red[w];
    atomicAdd(out, t * (1.0f / (float)N_ROWS));
  }
}

// ---------------------------------------------------------------------------
extern "C" void kernel_launch(void* const* d_in, const int* in_sizes, int n_in,
                              void* d_out, int out_size, void* d_ws, size_t ws_size,
                              hipStream_t stream) {
  const float* feat  = (const float*)d_in[0];
  const int*   label = (const int*)d_in[1];
  float*       out   = (float*)d_out;

  // Workspace layout: f16 normalized matrix, then per-row sumP / sumN.
  _Float16* fH   = (_Float16*)d_ws;
  float*    sums = (float*)((char*)d_ws + (size_t)N_ROWS * K_DIM * sizeof(_Float16));
  float*    sumP = sums;
  float*    sumN = sums + N_ROWS;

  init_kernel<<<(2 * N_ROWS + 255) / 256, 256, 0, stream>>>(sums, out);
  normalize_kernel<<<N_ROWS / 8, 256, 0, stream>>>(feat, fH);

  dim3 grid(N_ROWS / ROWS_PER_BLK, N_ROWS / COLS_PER_BLK);  // 128 x 8
  size_t shmem = (size_t)(ROWS_PER_BLK + 2 * COLS_PER_TILE) * LDS_K *
                 sizeof(_Float16);  // ~195 KB dynamic LDS
  cosface_main_kernel<<<grid, 256, shmem, stream>>>(fH, label, sumP, sumN);

  finalize_kernel<<<N_ROWS / 256, 256, 0, stream>>>(sumP, sumN, out);
}
